// UniMatchRNN_3109556322499
// MI455X (gfx1250) — matrix operand order repaired and checked by reading, not compiled, not run on hardware
//
#include <hip/hip_runtime.h>
#include <hip/hip_bf16.h>

// Problem dims (match reference)
#define L_   512
#define Q_   64
#define B_   64
#define D_   256
#define H_   256
#define RIN_ 512
#define G4_  1024

// Persistent-kernel config: 32 blocks x 256 threads = 256 wave32s
#define NB_   32
#define TPB_  256
#define NT_   (NB_ * TPB_)

typedef __attribute__((ext_vector_type(16))) __bf16 v16bf;
typedef __attribute__((ext_vector_type(8)))  float  v8f;
typedef __attribute__((ext_vector_type(4)))  unsigned u32x4;
typedef __attribute__((ext_vector_type(8)))  unsigned u32x8;

union FragBF { v16bf v; unsigned u[8]; };
union AccF   { v8f  v; float    f[8]; };

#define LOG2E 1.4426950408889634f

__device__ __forceinline__ unsigned short f2bf(float x) {
  unsigned u = __float_as_uint(x);
  u += 0x7fffu + ((u >> 16) & 1u);   // round-to-nearest-even
  return (unsigned short)(u >> 16);
}

__device__ __forceinline__ float fexp2(float x) { return __builtin_amdgcn_exp2f(x); }

__device__ __forceinline__ float fsigmoid(float x) {
  float e = fexp2(fminf(fmaxf(x, -30.f), 30.f) * LOG2E);   // e^x
  return e / (e + 1.0f);
}
__device__ __forceinline__ float ftanh_(float x) {
  float xx = fminf(fmaxf(x, -15.f), 15.f);
  float e = fexp2(xx * (2.0f * LOG2E));                    // e^{2x}
  return (e - 1.0f) / (e + 1.0f);
}

// ---- CDNA5 async / tensor data movement (inline asm; portable across toolchains) ----

// LDS byte address of a pointer derived from a __shared__ array: per the CDNA5
// aperture mapping, the low 32 bits of the generic address ARE the LDS address.
__device__ __forceinline__ unsigned lds_addr_of(const void* p) {
  return (unsigned)(size_t)p;
}

// Cooperative block-wide async copy global -> LDS (ASYNCcnt path), 16B per op.
__device__ __forceinline__ void stage_async(unsigned short* lds, const void* g, int bytes) {
  unsigned base = lds_addr_of(lds);
  const char* gs = (const char*)g;
  for (int o = threadIdx.x * 16; o < bytes; o += TPB_ * 16) {
    asm volatile("global_load_async_to_lds_b128 %0, %1, off"
                 :: "v"(base + (unsigned)o), "v"(gs + o) : "memory");
  }
  asm volatile("s_wait_asynccnt 0x0" ::: "memory");
  __syncthreads();
}

// Tensor Data Mover: 1-row 2D descriptor copying `nelem` bf16 elements global->LDS.
// D# layout per CDNA5 ISA ch.8 (group0: count/lds/global/type, group1: dims/strides).
__device__ __forceinline__ void tdm_load_row(unsigned short* lds, const void* g,
                                             unsigned nelem) {
  unsigned ldsa = lds_addr_of(lds);
  unsigned long long ga = (unsigned long long)(size_t)g;
  u32x4 g0 = {
      1u,                                                    // count=1, user D#
      ldsa,                                                  // lds_addr
      (unsigned)(ga & 0xffffffffu),                          // global_addr[31:0]
      (unsigned)((ga >> 32) & 0x1ffffffu) | (2u << 30)       // addr[56:32] | type=2
  };
  u32x8 g1 = {
      (1u << 16),                       // data_size=1 (2B), no multicast/pad/iterate
      (nelem & 0xffffu) << 16,          // tensor_dim0[15:0]
      ((nelem >> 16) & 0xffffu) | (1u << 16),  // tensor_dim0[31:16] | tensor_dim1=1
      (nelem & 0xffffu) << 16,          // tile_dim0
      1u,                               // tile_dim1=1, tile_dim2 unused
      nelem,                            // tensor_dim0_stride[31:0]
      0u, 0u                            // stride hi / tensor_dim1_stride
  };
  asm volatile("tensor_load_to_lds %0, %1" :: "s"(g0), "s"(g1) : "memory");
}

// ---- WMMA fragment loaders (layouts per CDNA5 ISA 7.12.2, wave32) ----

// A 16x32 bf16, rows row0..row0+15 of a row-major [rows x ldk] bf16 matrix.
// Works for both global and LDS-resident (shared-array) sources after inlining.
__device__ __forceinline__ v16bf load_a_bf(const unsigned short* A, int row0, int ldk,
                                           int k0, int lane) {
  FragBF f;
  const unsigned* Ar = (const unsigned*)(A + (size_t)(row0 + (lane & 15)) * ldk);
  int kb = (lane >> 4) * 8;
#pragma unroll
  for (int v = 0; v < 8; ++v) {
    int k = k0 + ((v >> 2) * 16) + kb + 2 * (v & 3);
    f.u[v] = Ar[k >> 1];
  }
  return f.v;
}

// Same A fragment but converting fp32 source on the fly.
__device__ __forceinline__ v16bf load_a_f32(const float* A, int row0, int ldk,
                                            int k0, int lane) {
  FragBF f;
  const float* Ar = A + (size_t)(row0 + (lane & 15)) * ldk;
  int kb = (lane >> 4) * 8;
#pragma unroll
  for (int v = 0; v < 8; ++v) {
    int k = k0 + ((v >> 2) * 16) + kb + 2 * (v & 3);
    unsigned lo = f2bf(Ar[k]);
    unsigned hi = f2bf(Ar[k + 1]);
    f.u[v] = lo | (hi << 16);
  }
  return f.v;
}

// B 32x16 bf16 from a TRANSPOSED weight (stored [n][k], k contiguous) -> dword loads.
__device__ __forceinline__ v16bf load_b_bf(const unsigned short* Wt, int n0, int ldk,
                                           int k0, int lane) {
  FragBF f;
  const unsigned* Wr = (const unsigned*)(Wt + (size_t)(n0 + (lane & 15)) * ldk);
  int kb = (lane >> 4) * 8;
#pragma unroll
  for (int v = 0; v < 8; ++v) {
    int k = k0 + ((v >> 2) * 16) + kb + 2 * (v & 3);
    f.u[v] = Wr[k >> 1];
  }
  return f.v;
}

#define WMMA_BF16(acc, a, b)                                                     \
  (acc) = __builtin_amdgcn_wmma_f32_16x16x32_bf16(false, (a), false, (b),        \
                                                  (short)0, (acc), false, false)

// ---- Prep kernels ----

__global__ void cast_transpose_kernel(const float* __restrict__ src,
                                      unsigned short* __restrict__ dst, int K, int N) {
  int idx = blockIdx.x * blockDim.x + threadIdx.x;
  int total = K * N;
  for (; idx < total; idx += gridDim.x * blockDim.x) {
    int k = idx / N, n = idx - k * N;
    dst[(size_t)n * K + k] = f2bf(src[idx]);
  }
}

__global__ void zero_state_kernel(float* h, float* c, unsigned short* h_bf,
                                  unsigned* bar) {
  int idx = blockIdx.x * blockDim.x + threadIdx.x;
  if (idx < B_ * H_) { h[idx] = 0.f; c[idx] = 0.f; h_bf[idx] = 0; }
  if (idx < 2) bar[idx] = 0;
}

// out[M x N] = A[M x K] @ W[K x N] + bias; one 16x16 C tile per wave32 (WMMA bf16).
__global__ void proj_gemm_kernel(const float* __restrict__ A,
                                 const unsigned short* __restrict__ Wt,
                                 const float* __restrict__ bias,
                                 float* __restrict__ out, int M, int N, int K) {
  int lane = threadIdx.x & 31;
  int wave = (blockIdx.x * blockDim.x + threadIdx.x) >> 5;
  int ntn = N >> 4;
  int total = (M >> 4) * ntn;
  if (wave >= total) return;
  int tm = wave / ntn, tn = wave - tm * ntn;

  AccF acc;
#pragma unroll
  for (int i = 0; i < 8; ++i) acc.f[i] = 0.f;

  for (int k0 = 0; k0 < K; k0 += 32) {
    v16bf a = load_a_f32(A, tm * 16, K, k0, lane);
    v16bf b = load_b_bf(Wt, tn * 16, K, k0, lane);
    WMMA_BF16(acc.v, a, b);
  }
  int n  = tn * 16 + (lane & 15);
  int mb = tm * 16 + ((lane >> 4) * 8);
  float bb = bias[n];
#pragma unroll
  for (int r = 0; r < 8; ++r)
    out[(size_t)(mb + r) * N + n] = acc.f[r] + bb;
}

// ---- Persistent recurrent kernel ----

__device__ void grid_sync(unsigned* bar) {
  __syncthreads();
  if (threadIdx.x == 0) {
    __threadfence();
    unsigned gen = atomicAdd(&bar[1], 0u);
    unsigned t   = atomicAdd(&bar[0], 1u);
    if (t == (unsigned)(NB_ - 1)) {
      bar[0] = 0;
      __threadfence();
      atomicAdd(&bar[1], 1u);
    } else {
      while (atomicAdd(&bar[1], 0u) == gen) __builtin_amdgcn_s_sleep(1);
    }
    __threadfence();
  }
  __syncthreads();
}

__global__ void __launch_bounds__(TPB_)
umrnn_recurrent_kernel(
    const float* __restrict__ Hp, const float* __restrict__ Hq,
    const float* __restrict__ Hq_mask,
    const float* __restrict__ Wg, const float* __restrict__ bg,
    const float* __restrict__ br, const float* __restrict__ bgate,
    const float* __restrict__ b_ih, const float* __restrict__ b_hh,
    const float* __restrict__ wq_hq, const float* __restrict__ wp_hp,
    const unsigned short* __restrict__ WrT, const unsigned short* __restrict__ WgateT,
    const unsigned short* __restrict__ WihT, const unsigned short* __restrict__ WhhT,
    float* __restrict__ h, float* __restrict__ c, unsigned short* __restrict__ h_bf,
    float* __restrict__ hWr, float* __restrict__ scores_p, float* __restrict__ alpha,
    float* __restrict__ z, unsigned short* __restrict__ z_bf,
    unsigned short* __restrict__ z2_bf, float* __restrict__ gates,
    unsigned* __restrict__ bar,
    float* __restrict__ out_h, float* __restrict__ out_alpha,
    float* __restrict__ out_gate)
{
  __shared__ unsigned short stageA[32768];   // 64KB LDS A-operand staging buffer

  const int tid  = blockIdx.x * TPB_ + threadIdx.x;
  const int lane = threadIdx.x & 31;
  const int gw   = tid >> 5;   // global wave id, [0,256)

  for (int t = 0; t < L_; ++t) {
    // ---- Phase 1: hWr = h @ Wr + br  (64x256); h staged in LDS via async copy ----
    if (blockIdx.x < 8) {                       // gw in [0,64): one tile per wave
      stage_async(stageA, h_bf, B_ * H_ * 2);   // 32KB
      int tm = gw & 3, tn = gw >> 2;
      AccF acc;
#pragma unroll
      for (int i = 0; i < 8; ++i) acc.f[i] = 0.f;
      for (int k0 = 0; k0 < H_; k0 += 32) {
        v16bf a = load_a_bf(stageA, tm * 16, H_, k0, lane);   // ds reads
        v16bf b = load_b_bf(WrT, tn * 16, H_, k0, lane);
        WMMA_BF16(acc.v, a, b);
      }
      int n  = tn * 16 + (lane & 15);
      int mb = tm * 16 + ((lane >> 4) * 8);
      float bb = br[n];
#pragma unroll
      for (int r = 0; r < 8; ++r)
        hWr[(mb + r) * H_ + n] = acc.f[r] + bb;
    }
    grid_sync(bar);

    // ---- Phase 2: partial scores; all 8192 threads, 2-way split of the H sum ----
    {
      int p = tid & 4095;          // (b,q) pair
      int half = tid >> 12;        // 0 or 1: which half of H
      int b = p >> 6, q = p & 63;
      const float* wq = wq_hq + ((size_t)q * B_ + b) * H_ + half * 128;
      const float* wp = wp_hp + ((size_t)t * B_ + b) * H_ + half * 128;
      const float* hw = hWr + b * H_ + half * 128;
      const float* wg = Wg + half * 128;
      float s = 0.f;
      for (int hh = 0; hh < 128; ++hh)
        s += ftanh_(wq[hh] + wp[hh] + hw[hh]) * wg[hh];
      scores_p[half * 4096 + p] = s;
    }
    grid_sync(bar);

    // ---- Phase 3: masked softmax over Q, one wave per batch row ----
    if (gw < B_) {
      int b = gw;
      float bg0 = bg[0];
      int i0 = b * Q_ + lane, i1 = i0 + 32;
      float x0 = scores_p[i0] + scores_p[4096 + i0] + bg0;
      float x1 = scores_p[i1] + scores_p[4096 + i1] + bg0;
      float mx = fmaxf(x0, x1);
#pragma unroll
      for (int off = 16; off > 0; off >>= 1)
        mx = fmaxf(mx, __shfl_xor(mx, off, 32));
      float m0 = Hq_mask[b * Q_ + lane];
      float m1 = Hq_mask[b * Q_ + lane + 32];
      float e0 = fexp2((x0 - mx) * LOG2E) * m0;
      float e1 = fexp2((x1 - mx) * LOG2E) * m1;
      float sm = e0 + e1;
#pragma unroll
      for (int off = 16; off > 0; off >>= 1)
        sm += __shfl_xor(sm, off, 32);
      float inv = 1.0f / (sm + 1e-6f);
      float a0 = e0 * inv, a1 = e1 * inv;
      alpha[b * Q_ + lane]      = a0;
      alpha[b * Q_ + lane + 32] = a1;
      out_alpha[((size_t)b * Q_ + lane) * L_ + t]      = a0;
      out_alpha[((size_t)b * Q_ + lane + 32) * L_ + t] = a1;
    }
    grid_sync(bar);

    // ---- Phase 4: qa = alpha @ Hq ; z = [Hp_t, qa]  (fp32 + bf16 copies) ----
    for (int idx = tid; idx < B_ * D_; idx += NT_) {
      int b = idx >> 8, d = idx & 255;
      const float* al = alpha + b * Q_;
      float qa = 0.f;
#pragma unroll 4
      for (int q = 0; q < Q_; ++q)
        qa += al[q] * Hq[((size_t)q * B_ + b) * D_ + d];
      float hp = Hp[((size_t)t * B_ + b) * D_ + d];
      z[b * RIN_ + d]        = hp;
      z[b * RIN_ + D_ + d]   = qa;
      z_bf[b * RIN_ + d]      = f2bf(hp);
      z_bf[b * RIN_ + D_ + d] = f2bf(qa);
    }
    grid_sync(bar);

    // ---- Phase 5: gate = sigmoid(z @ Wgate + bgate); z2 = gate*z ; z in LDS ----
    if (blockIdx.x < 16) {                      // gw in [0,128): one tile per wave
      stage_async(stageA, z_bf, B_ * RIN_ * 2); // 64KB
      int tm = gw & 3, tn = gw >> 2;
      AccF acc;
#pragma unroll
      for (int i = 0; i < 8; ++i) acc.f[i] = 0.f;
      for (int k0 = 0; k0 < RIN_; k0 += 32) {
        v16bf a = load_a_bf(stageA, tm * 16, RIN_, k0, lane);
        v16bf b = load_b_bf(WgateT, tn * 16, RIN_, k0, lane);
        WMMA_BF16(acc.v, a, b);
      }
      int n  = tn * 16 + (lane & 15);
      int mb = tm * 16 + ((lane >> 4) * 8);
      float bb = bgate[n];
#pragma unroll
      for (int r = 0; r < 8; ++r) {
        int b = mb + r;
        float gate = fsigmoid(acc.f[r] + bb);
        float z2 = gate * z[b * RIN_ + n];
        z2_bf[b * RIN_ + n] = f2bf(z2);
        out_gate[((size_t)b * RIN_ + n) * L_ + t] = gate;
      }
    }
    grid_sync(bar);

    // ---- Phase 6: gates = z2 @ W_ih + h @ W_hh + biases ; z2 staged via TDM ----
    if ((threadIdx.x >> 5) == 0) {
      tdm_load_row(stageA, z2_bf, B_ * RIN_);   // 64KB via Tensor Data Mover
      __builtin_amdgcn_s_wait_tensorcnt(0);
    }
    __syncthreads();
    {
      int tm = gw & 3, tn = gw >> 2;   // 256 tiles, exactly one per wave
      AccF acc;
#pragma unroll
      for (int i = 0; i < 8; ++i) acc.f[i] = 0.f;
      for (int k0 = 0; k0 < RIN_; k0 += 32) {
        v16bf a = load_a_bf(stageA, tm * 16, RIN_, k0, lane);   // LDS
        v16bf b = load_b_bf(WihT, tn * 16, RIN_, k0, lane);
        WMMA_BF16(acc.v, a, b);
      }
      for (int k0 = 0; k0 < H_; k0 += 32) {
        v16bf a = load_a_bf(h_bf, tm * 16, H_, k0, lane);       // global (L0-hot)
        v16bf b = load_b_bf(WhhT, tn * 16, H_, k0, lane);
        WMMA_BF16(acc.v, a, b);
      }
      int n  = tn * 16 + (lane & 15);
      int mb = tm * 16 + ((lane >> 4) * 8);
      float bb = b_ih[n] + b_hh[n];
#pragma unroll
      for (int r = 0; r < 8; ++r)
        gates[(mb + r) * G4_ + n] = acc.f[r] + bb;
    }
    grid_sync(bar);

    // ---- Phase 7: LSTM pointwise; refresh h, c, h_bf; emit result[t] ----
    for (int idx = tid; idx < B_ * H_; idx += NT_) {
      int b = idx >> 8, hh = idx & 255;
      const float* g = gates + b * G4_;
      float gi = fsigmoid(g[hh]);
      float gf = fsigmoid(g[H_ + hh]);
      float gg = ftanh_(g[2 * H_ + hh]);
      float go = fsigmoid(g[3 * H_ + hh]);
      float cn = gf * c[idx] + gi * gg;
      float hn = go * ftanh_(cn);
      c[idx] = cn;
      h[idx] = hn;
      h_bf[idx] = f2bf(hn);
      out_h[(size_t)t * (B_ * H_) + idx] = hn;
    }
    grid_sync(bar);
  }
}

// ---- Host launcher ----

extern "C" void kernel_launch(void* const* d_in, const int* in_sizes, int n_in,
                              void* d_out, int out_size, void* d_ws, size_t ws_size,
                              hipStream_t stream) {
  (void)in_sizes; (void)n_in; (void)out_size; (void)ws_size;

  const float* Hp      = (const float*)d_in[0];
  const float* Hq      = (const float*)d_in[1];
  const float* Hq_mask = (const float*)d_in[2];
  const float* Wq      = (const float*)d_in[3];
  const float* bq      = (const float*)d_in[4];
  const float* Wp      = (const float*)d_in[5];
  const float* bp      = (const float*)d_in[6];
  const float* Wr      = (const float*)d_in[7];
  const float* br      = (const float*)d_in[8];
  const float* Wg      = (const float*)d_in[9];
  const float* bg      = (const float*)d_in[10];
  const float* Wgate   = (const float*)d_in[11];
  const float* bgate   = (const float*)d_in[12];
  const float* W_ih    = (const float*)d_in[13];
  const float* W_hh    = (const float*)d_in[14];
  const float* b_ih    = (const float*)d_in[15];
  const float* b_hh    = (const float*)d_in[16];

  char* wsb = (char*)d_ws;
  size_t off = 0;
  auto take = [&](size_t bytes) -> char* {
    char* p = wsb + off;
    off = (off + bytes + 255) & ~(size_t)255;
    return p;
  };

  float* wq_hq = (float*)take((size_t)Q_ * B_ * H_ * 4);
  float* wp_hp = (float*)take((size_t)L_ * B_ * H_ * 4);
  unsigned short* WqT    = (unsigned short*)take((size_t)D_ * H_ * 2);
  unsigned short* WpT    = (unsigned short*)take((size_t)D_ * H_ * 2);
  unsigned short* WrT    = (unsigned short*)take((size_t)H_ * H_ * 2);
  unsigned short* WgateT = (unsigned short*)take((size_t)RIN_ * RIN_ * 2);
  unsigned short* WihT   = (unsigned short*)take((size_t)RIN_ * G4_ * 2);
  unsigned short* WhhT   = (unsigned short*)take((size_t)H_ * G4_ * 2);
  float* h     = (float*)take((size_t)B_ * H_ * 4);
  float* c     = (float*)take((size_t)B_ * H_ * 4);
  unsigned short* h_bf  = (unsigned short*)take((size_t)B_ * H_ * 2);
  float* hWr   = (float*)take((size_t)B_ * H_ * 4);
  float* scores_p = (float*)take((size_t)2 * B_ * Q_ * 4);
  float* alpha    = (float*)take((size_t)B_ * Q_ * 4);
  float* z     = (float*)take((size_t)B_ * RIN_ * 4);
  unsigned short* z_bf  = (unsigned short*)take((size_t)B_ * RIN_ * 2);
  unsigned short* z2_bf = (unsigned short*)take((size_t)B_ * RIN_ * 2);
  float* gates = (float*)take((size_t)B_ * G4_ * 4);
  unsigned* bar = (unsigned*)take(2 * sizeof(unsigned));

  // 1) Weight casts (transposed bf16)
  cast_transpose_kernel<<<256, 256, 0, stream>>>(Wq, WqT, D_, H_);
  cast_transpose_kernel<<<256, 256, 0, stream>>>(Wp, WpT, D_, H_);
  cast_transpose_kernel<<<256, 256, 0, stream>>>(Wr, WrT, H_, H_);
  cast_transpose_kernel<<<512, 256, 0, stream>>>(Wgate, WgateT, RIN_, RIN_);
  cast_transpose_kernel<<<1024, 256, 0, stream>>>(W_ih, WihT, RIN_, G4_);
  cast_transpose_kernel<<<512, 256, 0, stream>>>(W_hh, WhhT, H_, G4_);
  zero_state_kernel<<<(B_ * H_ + 255) / 256, 256, 0, stream>>>(h, c, h_bf, bar);

  // 2) Hoisted projections via WMMA (tile/wave; 8 waves per block)
  proj_gemm_kernel<<<((L_ * B_ / 16) * (H_ / 16)) / 8, 256, 0, stream>>>(
      Hp, WpT, bp, wp_hp, L_ * B_, H_, D_);
  proj_gemm_kernel<<<((Q_ * B_ / 16) * (H_ / 16)) / 8, 256, 0, stream>>>(
      Hq, WqT, bq, wq_hq, Q_ * B_, H_, D_);

  // 3) Persistent recurrent scan
  float* out_h     = (float*)d_out;
  float* out_alpha = out_h + (size_t)L_ * B_ * H_;
  float* out_gate  = out_alpha + (size_t)B_ * Q_ * L_;

  umrnn_recurrent_kernel<<<NB_, TPB_, 0, stream>>>(
      Hp, Hq, Hq_mask, Wg, bg, br, bgate, b_ih, b_hh,
      wq_hq, wp_hp, WrT, WgateT, WihT, WhhT,
      h, c, h_bf, hWr, scores_p, alpha, z, z_bf, z2_bf, gates, bar,
      out_h, out_alpha, out_gate);
}
